// BiLSTMClassifier_15281493639352
// MI455X (gfx1250) — matrix operand order repaired and checked by reading, compiled, not run on hardware
//
#include <hip/hip_runtime.h>
#include <hip/hip_bf16.h>

typedef __bf16 bf16_t;
typedef __attribute__((ext_vector_type(16))) __bf16 v16bf;
typedef __attribute__((ext_vector_type(8)))  __bf16 v8bf;
typedef __attribute__((ext_vector_type(8)))  float  v8f;

#define H_DIM 256
#define GATES 1024   // 4*H
#define SEQ   8192

__device__ __forceinline__ v8f wmma_bf16(v16bf a, v16bf b, v8f c) {
  return __builtin_amdgcn_wmma_f32_16x16x32_bf16(false, a, false, b,
                                                 (short)0, c, false, false);
}

// ---------------------------------------------------------------------------
// f32 -> bf16 conversion (grid-stride)
// ---------------------------------------------------------------------------
__global__ void convert_f32_bf16(const float* __restrict__ in,
                                 bf16_t* __restrict__ out, int n) {
  int i = blockIdx.x * blockDim.x + threadIdx.x;
  int stride = gridDim.x * blockDim.x;
  for (; i < n; i += stride) out[i] = (bf16_t)in[i];
}

// ---------------------------------------------------------------------------
// Pack a [K][N] f32 row-major matrix into WMMA B-fragment order (bf16):
// dst[tn][kb][lane][j] where (ISA 7.12.2, 16-bit B 32x16):
//   k = kb*32 + (j<8 ? (lane>>4)*8 + j : 16 + (lane>>4)*8 + (j-8)) ,  n = tn*16 + (lane&15)
// so a wave loads one contiguous v16bf (32 B) per lane per K-block.
// ---------------------------------------------------------------------------
__global__ void pack_b_f32(const float* __restrict__ src, bf16_t* __restrict__ dst,
                           int K, int N) {
  const int kbN = K >> 5;
  const int total = (N >> 4) * kbN * 512;
  int i = blockIdx.x * blockDim.x + threadIdx.x;
  const int stride = gridDim.x * blockDim.x;
  for (; i < total; i += stride) {
    const int j    =  i       & 15;
    const int lane = (i >> 4) & 31;
    const int rest =  i >> 9;
    const int kb   = rest % kbN;
    const int tn   = rest / kbN;
    const int k = kb * 32 + ((j < 8) ? ((lane >> 4) * 8 + j) : (16 + (lane >> 4) * 8 + (j - 8)));
    const int n = tn * 16 + (lane & 15);
    dst[i] = (bf16_t)src[(size_t)k * N + n];
  }
}

// ---------------------------------------------------------------------------
// C[M,N] = A[M,K] @ B[K,N] + bias[N].  A bf16 row-major, B pre-packed bf16,
// C f32.  Each wave computes a 32x64 block (2 M-tiles x 4 N-tiles) with
// 8 independent v_wmma_f32_16x16x32_bf16 accumulator chains.
// ---------------------------------------------------------------------------
template <int KBN>
__global__ __launch_bounds__(128)
void wmma_gemm_bias(const bf16_t* __restrict__ A, const bf16_t* __restrict__ Bp,
                    const float* __restrict__ bias, float* __restrict__ C,
                    int M, int N) {
  constexpr int K = KBN * 32;
  const int lane = threadIdx.x & 31;
  const int wid  = blockIdx.x * (blockDim.x >> 5) + (threadIdx.x >> 5);
  const int ngrp = N >> 6;                   // groups of 4 N-tiles
  const int mgrp = M >> 5;                   // groups of 2 M-tiles
  if (wid >= mgrp * ngrp) return;            // wave-uniform
  const int gm = wid / ngrp, gn = wid % ngrp;
  const int m0  = gm << 5;
  const int tn0 = gn << 2;
  const int hi = lane >> 4, row = lane & 15;

  const bf16_t* aptr = A + (size_t)(m0 + row) * K + hi * 8;
  const bf16_t* bptr = Bp + ((size_t)tn0 * KBN * 32 + lane) * 16;

  v8f acc[2][4] = {};
#pragma unroll 4
  for (int kb = 0; kb < KBN; ++kb) {
    v16bf a[2];
#pragma unroll
    for (int mi = 0; mi < 2; ++mi) {
      const bf16_t* arow = aptr + (size_t)mi * 16 * K;
      v8bf lo  = *(const v8bf*)(arow);
      v8bf hi8 = *(const v8bf*)(arow + 16);
#pragma unroll
      for (int j = 0; j < 8; ++j) { a[mi][j] = lo[j]; a[mi][j + 8] = hi8[j]; }
    }
#pragma unroll
    for (int ni = 0; ni < 4; ++ni) {
      const v16bf b = *(const v16bf*)(bptr + (size_t)ni * KBN * 512);
      acc[0][ni] = wmma_bf16(a[0], b, acc[0][ni]);
      acc[1][ni] = wmma_bf16(a[1], b, acc[1][ni]);
    }
    aptr += 32;        // next K-block within the A rows
    bptr += 512;       // next packed K-block within each B tile
  }
#pragma unroll
  for (int mi = 0; mi < 2; ++mi)
#pragma unroll
    for (int ni = 0; ni < 4; ++ni) {
      const int n = (tn0 + ni) * 16 + row;
      const float bv = bias[n];
      float* crow = C + (size_t)(m0 + mi * 16 + hi * 8) * N + n;
#pragma unroll
      for (int r = 0; r < 8; ++r) crow[(size_t)r * N] = acc[mi][ni][r] + bv;
    }
}

// ---------------------------------------------------------------------------
// Persistent bidirectional LSTM scan. grid.x = 2 (dir), block = 1024 (32 waves).
// Each wave owns two 16-wide N tiles of W_hh: tile 0's 8 B-fragments live in
// registers (64 VGPRs, under the 8-wave/SIMD budget), tile 1's are staged once
// into a 256 KB LDS array (WGP has 320 KB).  Steady-state inner loop is pure
// LDS + WMMA: 2 ds_load_b128 (A: h on lanes 0/16, others read a zeroed LDS
// patch -> no EXEC juggling) + 2 ds_load_b128 (B tile 1) + 2 WMMAs per K-block.
// Next timestep's xp row (4 KB) is double-buffered into LDS with
// global_load_async_to_lds_b32 / s_wait_asynccnt (gfx1250 async path).
// ---------------------------------------------------------------------------
__global__ __launch_bounds__(1024)
void lstm_scan(const float* __restrict__ xp        /* [2][S][1024], bias folded */,
               const bf16_t* __restrict__ whh_pack /* [2][64][8][32][16] */,
               float* __restrict__ h_out           /* [S][512] fwd||bwd */,
               int S) {
  const int dir  = blockIdx.x;
  const int tid  = threadIdx.x;
  const int lane = tid & 31;
  const int wave = tid >> 5;          // 0..31
  const int hi   = lane >> 4, row = lane & 15;

  __shared__ __align__(16) bf16_t sh_w[32][8][32][16];  // 256 KB: tile-1 frags
  __shared__ __align__(16) bf16_t sh_hbf[H_DIM];
  __shared__ __align__(16) bf16_t sh_zero[32];          // 64 B of zeros
  __shared__ float sh_c[H_DIM];
  __shared__ float sh_g[GATES];
  __shared__ float sh_xp[2][GATES];

  if (tid < H_DIM) { sh_hbf[tid] = (bf16_t)0.0f; sh_c[tid] = 0.f; }
  if (tid < 32) sh_zero[tid] = (bf16_t)0.0f;

  const float*  xp_d = xp + (size_t)dir * S * GATES;
  const bf16_t* wp   = whh_pack + (size_t)dir * 64 * 8 * 512
                     + (size_t)(wave * 2) * 8 * 512;

  // tile 0 -> registers (8 x v16bf = 64 VGPRs)
  v16bf w0[8];
#pragma unroll
  for (int kb = 0; kb < 8; ++kb)
    w0[kb] = *(const v16bf*)(wp + ((size_t)kb * 32 + lane) * 16);
  // tile 1 -> LDS (wave-private slice, no barrier needed)
#pragma unroll
  for (int kb = 0; kb < 8; ++kb)
    *(v16bf*)(&sh_w[wave][kb][lane][0]) =
        *(const v16bf*)(wp + (size_t)8 * 512 + ((size_t)kb * 32 + lane) * 16);

  // stage xp row for the first timestep into buffer 0
  {
    const int s0 = dir ? (S - 1) : 0;
    unsigned lds = (unsigned)(uintptr_t)(&sh_xp[0][tid]);
    unsigned long long ga =
        (unsigned long long)(uintptr_t)(xp_d + (size_t)s0 * GATES + tid);
    asm volatile("global_load_async_to_lds_b32 %0, %1, off"
                 :: "v"(lds), "v"(ga) : "memory");
    asm volatile("s_wait_asynccnt 0x0" ::: "memory");
  }
  __syncthreads();

  for (int t = 0; t < S; ++t) {
    const int s   = dir ? (S - 1 - t) : t;
    const int buf = t & 1;
    // kick off async staging of the next timestep's xp row
    if (t + 1 < S) {
      const int sn = dir ? (s - 1) : (s + 1);
      unsigned lds = (unsigned)(uintptr_t)(&sh_xp[buf ^ 1][tid]);
      unsigned long long ga =
          (unsigned long long)(uintptr_t)(xp_d + (size_t)sn * GATES + tid);
      asm volatile("global_load_async_to_lds_b32 %0, %1, off"
                   :: "v"(lds), "v"(ga) : "memory");
    }
    // g = h @ W_hh : each wave owns two 16-wide N tiles (64 tiles = 1024 gates)
    v8f acc0 = {}, acc1 = {};
#pragma unroll
    for (int kb = 0; kb < 8; ++kb) {
      // A fragment: lanes with row==0 read h_prev, others read zeros
      const bf16_t* abase = (row == 0) ? (sh_hbf + kb * 32 + hi * 8) : sh_zero;
      v8bf lo  = *(const v8bf*)(abase);
      v8bf hi8 = *(const v8bf*)(abase + 16);
      v16bf a;
#pragma unroll
      for (int j = 0; j < 8; ++j) { a[j] = lo[j]; a[j + 8] = hi8[j]; }
      const v16bf b1 = *(const v16bf*)(&sh_w[wave][kb][lane][0]);
      acc0 = wmma_bf16(a, w0[kb], acc0);
      acc1 = wmma_bf16(a, b1, acc1);
    }
    // D row 0 lives in acc[0] of lanes 0..15
    if (lane < 16) {
      const int n0 = wave * 32 + lane;
      sh_g[n0]      = acc0[0] + sh_xp[buf][n0];
      sh_g[n0 + 16] = acc1[0] + sh_xp[buf][n0 + 16];
    }
    __syncthreads();
    if (tid < H_DIM) {
      const float iv = 1.f / (1.f + __expf(-sh_g[tid]));
      const float fv = 1.f / (1.f + __expf(-sh_g[tid + 256]));
      const float gv = tanhf(sh_g[tid + 512]);
      const float ov = 1.f / (1.f + __expf(-sh_g[tid + 768]));
      const float cv = fv * sh_c[tid] + iv * gv;
      const float hv = ov * tanhf(cv);
      sh_c[tid]   = cv;
      sh_hbf[tid] = (bf16_t)hv;
      h_out[(size_t)s * (2 * H_DIM) + dir * H_DIM + tid] = hv;
    }
    asm volatile("s_wait_asynccnt 0x0" ::: "memory");
    __syncthreads();
  }
}

// ---------------------------------------------------------------------------
// Final head: feat = [h[S-1, 0:256] || h[0, 256:512]]; out = (feat@w1+b1)@w2+b2
// ---------------------------------------------------------------------------
__global__ void head_kernel(const float* __restrict__ h, const float* __restrict__ w1,
                            const float* __restrict__ b1, const float* __restrict__ w2,
                            const float* __restrict__ b2, float* __restrict__ out,
                            int S) {
  __shared__ float feat[512];
  __shared__ float hid[32];
  const int tid = threadIdx.x;              // 64 threads
  for (int i = tid; i < 512; i += blockDim.x)
    feat[i] = (i < 256) ? h[(size_t)(S - 1) * 512 + i] : h[i];
  __syncthreads();
  if (tid < 32) {
    float s = b1[tid];
    for (int k = 0; k < 512; ++k) s += feat[k] * w1[k * 32 + tid];
    hid[tid] = s;
  }
  __syncthreads();
  if (tid < 2) {
    float s = b2[tid];
    for (int j = 0; j < 32; ++j) s += hid[j] * w2[j * 2 + tid];
    out[tid] = s;
  }
}

// ---------------------------------------------------------------------------
extern "C" void kernel_launch(void* const* d_in, const int* in_sizes, int n_in,
                              void* d_out, int out_size, void* d_ws, size_t ws_size,
                              hipStream_t stream) {
  const float* x     = (const float*)d_in[0];   // [8192,1024]
  const float* w_ih0 = (const float*)d_in[1];   // [2,1024,1024]
  const float* w_hh0 = (const float*)d_in[2];   // [2,256,1024]
  const float* b0    = (const float*)d_in[3];   // [2,1024]
  const float* w_ih  = (const float*)d_in[4];   // [4,2,512,1024]
  const float* w_hh  = (const float*)d_in[5];   // [4,2,256,1024]
  const float* b     = (const float*)d_in[6];   // [4,2,1024]
  const float* w1    = (const float*)d_in[7];   // [512,32]
  const float* b1    = (const float*)d_in[8];   // [32]
  const float* w2    = (const float*)d_in[9];   // [32,2]
  const float* b2    = (const float*)d_in[10];  // [2]
  (void)in_sizes; (void)n_in; (void)out_size; (void)ws_size;

  char* ws = (char*)d_ws;
  size_t off = 0;
  auto alloc = [&](size_t bytes) -> void* {
    void* p = ws + off;
    off = (off + bytes + 255) & ~(size_t)255;
    return p;
  };
  const size_t PK_IH0 = (size_t)64 * 32 * 512;    // per-dir pack elems (K=1024)
  const size_t PK_IH  = (size_t)64 * 16 * 512;    // per (l,d)        (K=512)
  const size_t PK_HH  = (size_t)64 *  8 * 512;    // per (layer,dir)  (K=256)

  bf16_t* x_bf    = (bf16_t*)alloc((size_t)SEQ * 1024 * 2);
  bf16_t* wih0_pk = (bf16_t*)alloc(2 * PK_IH0 * 2);
  bf16_t* wih_pk  = (bf16_t*)alloc(8 * PK_IH * 2);
  bf16_t* whh_pk  = (bf16_t*)alloc(10 * PK_HH * 2);
  float*  xp      = (float*) alloc((size_t)2 * SEQ * 1024 * 4);
  float*  h_f32   = (float*) alloc((size_t)SEQ * 512 * 4);
  bf16_t* hin_bf  = (bf16_t*)alloc((size_t)SEQ * 512 * 2);

  // --- one-time conversions / packing ---
  convert_f32_bf16<<<2048, 256, 0, stream>>>(x, x_bf, SEQ * 1024);
  for (int d = 0; d < 2; ++d) {
    pack_b_f32<<<2048, 256, 0, stream>>>(w_ih0 + (size_t)d * 1024 * 1024,
                                         wih0_pk + d * PK_IH0, 1024, 1024);
    pack_b_f32<<<1024, 256, 0, stream>>>(w_hh0 + (size_t)d * 256 * 1024,
                                         whh_pk + d * PK_HH, 256, 1024);
  }
  for (int l = 0; l < 4; ++l)
    for (int d = 0; d < 2; ++d) {
      pack_b_f32<<<2048, 256, 0, stream>>>(
          w_ih + (size_t)(l * 2 + d) * 512 * 1024,
          wih_pk + (size_t)(l * 2 + d) * PK_IH, 512, 1024);
      pack_b_f32<<<1024, 256, 0, stream>>>(
          w_hh + (size_t)(l * 2 + d) * 256 * 1024,
          whh_pk + (size_t)((l + 1) * 2 + d) * PK_HH, 256, 1024);
    }

  // 256 M-groups x 16 N-groups = 4096 waves, 4 waves/block
  const int gemmBlocks = (256 * 16) / 4;

  // --- layer 0: input projections + scan ---
  for (int d = 0; d < 2; ++d)
    wmma_gemm_bias<32><<<gemmBlocks, 128, 0, stream>>>(
        x_bf, wih0_pk + (size_t)d * PK_IH0, b0 + (size_t)d * 1024,
        xp + (size_t)d * SEQ * 1024, SEQ, 1024);
  lstm_scan<<<2, 1024, 0, stream>>>(xp, whh_pk, h_f32, SEQ);

  // --- layers 1..4 ---
  for (int l = 0; l < 4; ++l) {
    convert_f32_bf16<<<2048, 256, 0, stream>>>(h_f32, hin_bf, SEQ * 512);
    for (int d = 0; d < 2; ++d)
      wmma_gemm_bias<16><<<gemmBlocks, 128, 0, stream>>>(
          hin_bf, wih_pk + (size_t)(l * 2 + d) * PK_IH,
          b + (size_t)(l * 2 + d) * 1024,
          xp + (size_t)d * SEQ * 1024, SEQ, 1024);
    lstm_scan<<<2, 1024, 0, stream>>>(xp, whh_pk + (size_t)(l + 1) * 2 * PK_HH,
                                      h_f32, SEQ);
  }

  // --- classifier head ---
  head_kernel<<<1, 64, 0, stream>>>(h_f32, w1, b1, w2, b2, (float*)d_out, SEQ);
}
